// MentalNet_58737972740327
// MI455X (gfx1250) — matrix operand order repaired and checked by reading, not compiled, not run on hardware
//
#include <hip/hip_runtime.h>
#include <math.h>

// ---------------- problem constants (match reference) ----------------
#define NNODES 100000
#define NREL   3
#define NEDGE  1600000
#define HIDC   64
#define DFLAT  129          // HID*2 + 1
#define KTOP   30
#define C1N    16
#define C2N    32
#define KSZ    5
#define NGROUP 64
#define POOL15 (KTOP/2)             // 15
#define CONV2T (POOL15 - KSZ + 1)   // 11
#define OUTPER (C2N * CONV2T)       // 352

typedef __attribute__((ext_vector_type(16))) __bf16 v16bf;
typedef __attribute__((ext_vector_type(8)))  __bf16 v8bf;
typedef __attribute__((ext_vector_type(8)))  float  v8f;
typedef __attribute__((ext_vector_type(4)))  int    v4i;

#define AS1 __attribute__((address_space(1)))
#define AS3 __attribute__((address_space(3)))

#if defined(__HIP_DEVICE_COMPILE__) && __has_builtin(__builtin_amdgcn_global_load_async_to_lds_b128)
#define HAVE_ASYNC 1
#else
#define HAVE_ASYNC 0
#endif

// ---------------- utility ----------------
__global__ void k_zero(float* p, int n) {
    int i = blockIdx.x * blockDim.x + threadIdx.x;
    if (i < n) p[i] = 0.0f;
}
__global__ void k_zeroi(int* p, int n) {
    int i = blockIdx.x * blockDim.x + threadIdx.x;
    if (i < n) p[i] = 0;
}

// f32 -> bf16 elementwise
__global__ void k_cvtbf(const float* __restrict__ src, __bf16* __restrict__ dst, long long n) {
    long long i = (long long)blockIdx.x * blockDim.x + threadIdx.x;
    if (i < n) dst[i] = (__bf16)src[i];
}

// deg[dst] += ew[e]
__global__ void k_deg(const int* __restrict__ ei, const float* __restrict__ ew,
                      float* __restrict__ deg) {
    int e = blockIdx.x * blockDim.x + threadIdx.x;
    if (e < NEDGE) atomicAdd(&deg[ei[NEDGE + e]], ew[e]);
}

// dinv = rsqrt(deg+1); snorm = dinv*dinv
__global__ void k_norm(const float* __restrict__ deg, float* __restrict__ dinv,
                       float* __restrict__ snorm) {
    int n = blockIdx.x * blockDim.x + threadIdx.x;
    if (n < NNODES) {
        float di = rsqrtf(deg[n] + 1.0f);
        dinv[n]  = di;
        snorm[n] = di * di;
    }
}

// enorm[e] = dinv[src]*ew*dinv[dst]
__global__ void k_enorm(const int* __restrict__ ei, const float* __restrict__ ew,
                        const float* __restrict__ dinv, float* __restrict__ enorm) {
    int e = blockIdx.x * blockDim.x + threadIdx.x;
    if (e < NEDGE) enorm[e] = dinv[ei[e]] * ew[e] * dinv[ei[NEDGE + e]];
}

// Bt[n*Kdim + k] = (bf16) W[k*64 + n]   (transpose + convert, W is [Kdim,64])
__global__ void k_prepB(const float* __restrict__ W, __bf16* __restrict__ Bt, int Kdim) {
    int i = blockIdx.x * blockDim.x + threadIdx.x;
    if (i < HIDC * Kdim) {
        int n = i / Kdim, k = i % Kdim;
        Bt[(size_t)n * Kdim + k] = (__bf16)W[(size_t)k * HIDC + n];
    }
}

// ---------------- WMMA GEMM: C[M,64] = A[M,Kdim](bf16 packed) x B[Kdim,64]
// B pre-transposed bf16 Bt[64][Kdim], staged in LDS (padded). One wave -> 16x64 tile.
__global__ void __launch_bounds__(256)
k_gemm_bf16wmma(const __bf16* __restrict__ A, const __bf16* __restrict__ Bt, int Kdim,
                float* __restrict__ C, int M) {
    __shared__ __bf16 ldsB[HIDC * (128 + 8)];      // worst case Kp = 136
    const int Kp = Kdim + 8;                       // pad: conflict-free ds_load_b128
    int tid = threadIdx.x;

    // cooperative stage: Bt [64, Kdim] -> LDS [64, Kp] in 16B chunks
    int cpr = Kdim >> 3;                           // 8 bf16 per chunk
    int nchunks = HIDC * cpr;
    for (int i = tid; i < nchunks; i += 256) {
        int rowb = i / cpr, ch = i - rowb * cpr;
        const __bf16* gp = Bt + (size_t)rowb * Kdim + ch * 8;
        __bf16* lp = &ldsB[rowb * Kp + ch * 8];
#if HAVE_ASYNC
        __builtin_amdgcn_global_load_async_to_lds_b128(
            (AS1 v4i*)gp, (AS3 v4i*)lp, 0, 0);
#else
        *(uint4*)lp = *(const uint4*)gp;
#endif
    }
#if HAVE_ASYNC
#if __has_builtin(__builtin_amdgcn_s_wait_asynccnt)
    __builtin_amdgcn_s_wait_asynccnt(0);
#else
    asm volatile("s_wait_asynccnt 0x0" ::: "memory");
#endif
#endif
    __syncthreads();

    int wave = tid >> 5;
    int lane = tid & 31;
    int m0 = (blockIdx.x * 8 + wave) * 16;
    if (m0 >= M) return;                           // uniform per wave after the barrier

    int rsel  = lane >> 4;                         // 0: lanes 0-15, 1: lanes 16-31
    int row   = m0 + (lane & 15);
    int kgrp  = rsel * 8;                          // A-frag K sub-offset (0 or 8)
    int koff2 = rsel * 8;                          // B-frag K sub-offset (0 or 8)
    const __bf16* arow = A + (size_t)row * Kdim;

    v8f acc[4];
#pragma unroll
    for (int n = 0; n < 4; ++n) acc[n] = (v8f){};

    for (int kb = 0; kb < Kdim; kb += 32) {
        // A fragment (ISA 16-bit A 16x32 layout): two aligned 16B loads
        v8bf a0 = *(const v8bf*)(arow + kb + kgrp);
        v8bf a1 = *(const v8bf*)(arow + kb + 16 + kgrp);
        v16bf a;
#pragma unroll
        for (int i = 0; i < 8; ++i) { a[i] = a0[i]; a[i + 8] = a1[i]; }
#pragma unroll
        for (int n = 0; n < 4; ++n) {
            const __bf16* bp = &ldsB[(n * 16 + (lane & 15)) * Kp + kb + koff2];
            v8bf b0 = *(const v8bf*)bp;
            v8bf b1 = *(const v8bf*)(bp + 16);
            v16bf b;
#pragma unroll
            for (int i = 0; i < 8; ++i) { b[i] = b0[i]; b[i + 8] = b1[i]; }
            acc[n] = __builtin_amdgcn_wmma_f32_16x16x32_bf16(
                false, a, false, b, (short)0, acc[n], false, false);
        }
    }

    // C/D layout: lanes 0-15 -> rows m0+v, lanes 16-31 -> rows m0+8+v; col = lane&15
    int rbase = m0 + rsel * 8;
#pragma unroll
    for (int v = 0; v < 8; ++v) {
#pragma unroll
        for (int n = 0; n < 4; ++n)
            C[(size_t)(rbase + v) * HIDC + n * 16 + (lane & 15)] = acc[n][v];
    }
}

// ---------------- edge aggregation: agg[dst,c] += enorm[e]*h[src,c] ----------------
__global__ void k_eagg(const int* __restrict__ ei, const float* __restrict__ enorm,
                       const float* __restrict__ h, float* __restrict__ agg) {
    long long gid = (long long)blockIdx.x * blockDim.x + threadIdx.x;
    if (gid >= (long long)NEDGE * HIDC) return;
    int e = (int)(gid >> 6);
    int c = (int)(gid & 63);
    atomicAdd(&agg[(size_t)ei[NEDGE + e] * HIDC + c],
              enorm[e] * h[(size_t)ei[e] * HIDC + c]);
}

// scalar (HID=1) edge aggregation for layer 3
__global__ void k_eagg1(const int* __restrict__ ei, const float* __restrict__ enorm,
                        const float* __restrict__ h, float* __restrict__ agg) {
    int e = blockIdx.x * blockDim.x + threadIdx.x;
    if (e < NEDGE) atomicAdd(&agg[ei[NEDGE + e]], enorm[e] * h[ei[e]]);
}

// h = tanh(agg + snorm*hlin + b) -> feat[:, colOff..colOff+63]; optional bf16 copy
__global__ void k_post(const float* __restrict__ agg, const float* __restrict__ hlin,
                       const float* __restrict__ snorm, const float* __restrict__ b,
                       float* __restrict__ feat, int colOff, __bf16* __restrict__ hbf) {
    long long i = (long long)blockIdx.x * blockDim.x + threadIdx.x;
    if (i >= (long long)NNODES * HIDC) return;
    int n = (int)(i >> 6);
    int c = (int)(i & 63);
    float v = tanhf(agg[i] + snorm[n] * hlin[i] + b[c]);
    feat[(size_t)n * DFLAT + colOff + c] = v;
    if (hbf) hbf[i] = (__bf16)v;
}

// hl3[n] = dot(feat[n, 64..127], W2)
__global__ void k_dot3(const float* __restrict__ feat, const float* __restrict__ W2,
                       float* __restrict__ hl3) {
    int n = blockIdx.x * blockDim.x + threadIdx.x;
    if (n >= NNODES) return;
    const float* f = feat + (size_t)n * DFLAT + HIDC;
    float s = 0.0f;
#pragma unroll
    for (int c = 0; c < HIDC; ++c) s += f[c] * W2[c];
    hl3[n] = s;
}

// feat[:,128] = tanh(agg1 + snorm*hl3 + b2)
__global__ void k_post3(const float* __restrict__ agg1, const float* __restrict__ hl3,
                        const float* __restrict__ snorm, const float* __restrict__ b2,
                        float* __restrict__ feat) {
    int n = blockIdx.x * blockDim.x + threadIdx.x;
    if (n >= NNODES) return;
    feat[(size_t)n * DFLAT + (DFLAT - 1)] = tanhf(agg1[n] + snorm[n] * hl3[n] + b2[0]);
}

// ---------------- group bucketing (relation-invariant, run once) ----------------
__global__ void k_gcount(const int* __restrict__ group, int* __restrict__ gcount) {
    int n = blockIdx.x * blockDim.x + threadIdx.x;
    if (n < NNODES) atomicAdd(&gcount[group[n]], 1);
}
__global__ void k_gscan(const int* __restrict__ gcount, int* __restrict__ gstart) {
    if (threadIdx.x == 0 && blockIdx.x == 0) {
        int s = 0;
        for (int g = 0; g < NGROUP; ++g) { gstart[g] = s; s += gcount[g]; }
        gstart[NGROUP] = s;
    }
}
__global__ void k_gscatter(const int* __restrict__ group, const int* __restrict__ gstart,
                           int* __restrict__ gcursor, int* __restrict__ glist) {
    int n = blockIdx.x * blockDim.x + threadIdx.x;
    if (n >= NNODES) return;
    int g = group[n];
    int pos = atomicAdd(&gcursor[g], 1);
    glist[gstart[g] + pos] = n;
}

// ---------------- sort-pool: per-group top-K by feat[:,128] (desc, tie -> lowest idx)
// One wave per group; scans only that group's member list. Selection is a max over
// distinct (score, index) keys, so it is independent of glist's scatter order.
__global__ void k_topk(const float* __restrict__ feat, const int* __restrict__ glist,
                       const int* __restrict__ gstart, float* __restrict__ pooled) {
    int g = blockIdx.x;
    int lane = threadIdx.x;
    int s0 = gstart[g];
    int cnt = gstart[g + 1] - s0;
    float prev_s = INFINITY;
    int   prev_i = -1;
    for (int k = 0; k < KTOP; ++k) {
        float bs = -INFINITY;
        int   bi = 0x7fffffff;
        for (int j = lane; j < cnt; j += 32) {
            int n = glist[s0 + j];
            float s = feat[(size_t)n * DFLAT + (DFLAT - 1)];
            bool after = (s < prev_s) || (s == prev_s && n > prev_i);
            if (!after) continue;
            if (s > bs || (s == bs && n < bi)) { bs = s; bi = n; }
        }
#pragma unroll
        for (int off = 16; off >= 1; off >>= 1) {
            float os = __shfl_xor(bs, off, 32);
            int   oi = __shfl_xor(bi, off, 32);
            if (os > bs || (os == bs && oi < bi)) { bs = os; bi = oi; }
        }
        prev_s = bs; prev_i = bi;
        int idx = (bi < NNODES) ? bi : 0;
        for (int c = lane; c < DFLAT; c += 32)
            pooled[((size_t)g * KTOP + k) * DFLAT + c] = feat[(size_t)idx * DFLAT + c];
    }
}

// ---------------- CNN head: conv1(129,s129)+relu -> maxpool2 -> conv2(5)+relu, += out
__global__ void k_cnn(const float* __restrict__ pooled,
                      const float* __restrict__ c1w, const float* __restrict__ c1b,
                      const float* __restrict__ c2w, const float* __restrict__ c2b,
                      float* __restrict__ out) {
    __shared__ float prow[KTOP * DFLAT];   // 3870
    __shared__ float yc[C1N * KTOP];       // 480
    __shared__ float y1[C1N * POOL15];     // 240
    int g = blockIdx.x;
    int tid = threadIdx.x;
    for (int i = tid; i < KTOP * DFLAT; i += blockDim.x)
        prow[i] = pooled[(size_t)g * KTOP * DFLAT + i];
    __syncthreads();
    for (int o = tid; o < C1N * KTOP; o += blockDim.x) {
        int c1 = o / KTOP, t = o % KTOP;
        float acc = c1b[c1];
        for (int j = 0; j < DFLAT; ++j)
            acc += c1w[c1 * DFLAT + j] * prow[t * DFLAT + j];
        yc[c1 * KTOP + t] = fmaxf(acc, 0.0f);
    }
    __syncthreads();
    for (int o = tid; o < C1N * POOL15; o += blockDim.x) {
        int c1 = o / POOL15, p = o % POOL15;
        y1[o] = fmaxf(yc[c1 * KTOP + 2 * p], yc[c1 * KTOP + 2 * p + 1]);
    }
    __syncthreads();
    for (int o = tid; o < OUTPER; o += blockDim.x) {
        int c2 = o / CONV2T, t = o % CONV2T;
        float acc = c2b[c2];
#pragma unroll
        for (int c1 = 0; c1 < C1N; ++c1)
#pragma unroll
            for (int j = 0; j < KSZ; ++j)
                acc += c2w[(c2 * C1N + c1) * KSZ + j] * y1[c1 * POOL15 + t + j];
        out[(size_t)g * OUTPER + o] += fmaxf(acc, 0.0f);
    }
}

// ---------------- orchestration ----------------
extern "C" void kernel_launch(void* const* d_in, const int* in_sizes, int n_in,
                              void* d_out, int out_size, void* d_ws, size_t ws_size,
                              hipStream_t stream) {
    const float* x     = (const float*)d_in[0];
    const int*   eidx  = (const int*)  d_in[1];
    const float* ew    = (const float*)d_in[2];
    const int*   group = (const int*)  d_in[3];
    const float* W0    = (const float*)d_in[4];
    const float* b0    = (const float*)d_in[5];
    const float* W1    = (const float*)d_in[6];
    const float* b1    = (const float*)d_in[7];
    const float* W2    = (const float*)d_in[8];
    const float* b2    = (const float*)d_in[9];
    const float* c1w   = (const float*)d_in[10];
    const float* c1b   = (const float*)d_in[11];
    const float* c2w   = (const float*)d_in[12];
    const float* c2b   = (const float*)d_in[13];
    float* out = (float*)d_out;

    char* ws = (char*)d_ws;
    size_t off = 0;
    auto alloc = [&](size_t bytes) -> void* {
        off = (off + 255) & ~(size_t)255;
        void* p = ws + off;
        off += bytes;
        return p;
    };
    float*  deg    = (float*) alloc((size_t)NNODES * 4);
    float*  dinv   = (float*) alloc((size_t)NNODES * 4);
    float*  snorm  = (float*) alloc((size_t)NNODES * 4);
    float*  enorm  = (float*) alloc((size_t)NEDGE * 4);
    float*  hlin   = (float*) alloc((size_t)NNODES * HIDC * 4);
    float*  agg    = (float*) alloc((size_t)NNODES * HIDC * 4);
    float*  feat   = (float*) alloc((size_t)NNODES * DFLAT * 4);
    float*  hl3    = (float*) alloc((size_t)NNODES * 4);
    float*  agg1   = (float*) alloc((size_t)NNODES * 4);
    float*  pooled = (float*) alloc((size_t)NGROUP * KTOP * DFLAT * 4);
    __bf16* xbf    = (__bf16*)alloc((size_t)NNODES * 128 * 2);
    __bf16* h1bf   = (__bf16*)alloc((size_t)NNODES * HIDC * 2);
    __bf16* Bt0    = (__bf16*)alloc((size_t)HIDC * 128 * 2);
    __bf16* Bt1    = (__bf16*)alloc((size_t)HIDC * HIDC * 2);
    int*    gcount = (int*)   alloc((size_t)NGROUP * 4);
    int*    gstart = (int*)   alloc((size_t)(NGROUP + 1) * 4);
    int*    gcursor= (int*)   alloc((size_t)NGROUP * 4);
    int*    glist  = (int*)   alloc((size_t)NNODES * 4);

    const int T = 256;
    auto cdiv = [](long long a, long long b) { return (unsigned)((a + b - 1) / b); };

    const unsigned gN    = cdiv(NNODES, T);
    const unsigned gE    = cdiv(NEDGE, T);
    const unsigned gNH   = cdiv((long long)NNODES * HIDC, T);
    const unsigned gEH   = cdiv((long long)NEDGE * HIDC, T);
    const unsigned gGEMM = cdiv(NNODES / 16, 8);    // 8 waves / 256-thread block

    // one-time prep: zero output, bucket nodes by group, x -> bf16
    k_zero   <<<cdiv(NGROUP * OUTPER, T), T, 0, stream>>>(out, NGROUP * OUTPER);
    k_zeroi  <<<1, 64, 0, stream>>>(gcount, NGROUP);
    k_gcount <<<gN, T, 0, stream>>>(group, gcount);
    k_gscan  <<<1, 1, 0, stream>>>(gcount, gstart);
    k_zeroi  <<<1, 64, 0, stream>>>(gcursor, NGROUP);
    k_gscatter<<<gN, T, 0, stream>>>(group, gstart, gcursor, glist);
    k_cvtbf  <<<cdiv((long long)NNODES * 128, T), T, 0, stream>>>(x, xbf, (long long)NNODES * 128);

    for (int r = 0; r < NREL; ++r) {
        const int*   ei  = eidx + (size_t)r * 2 * NEDGE;
        const float* ewr = ew   + (size_t)r * NEDGE;

        // GCN normalization
        k_zero <<<gN, T, 0, stream>>>(deg, NNODES);
        k_deg  <<<gE, T, 0, stream>>>(ei, ewr, deg);
        k_norm <<<gN, T, 0, stream>>>(deg, dinv, snorm);
        k_enorm<<<gE, T, 0, stream>>>(ei, ewr, dinv, enorm);

        // weights -> bf16 transposed
        k_prepB<<<cdiv(HIDC * 128, T), T, 0, stream>>>(W0 + (size_t)r * 128 * HIDC, Bt0, 128);
        k_prepB<<<cdiv(HIDC * HIDC, T), T, 0, stream>>>(W1 + (size_t)r * HIDC * HIDC, Bt1, HIDC);

        // layer 1: hlin = x @ W0 (WMMA, B in LDS), aggregate, tanh -> feat[:,0:64] + h1bf
        k_gemm_bf16wmma<<<gGEMM, T, 0, stream>>>(xbf, Bt0, 128, hlin, NNODES);
        k_zero<<<gNH, T, 0, stream>>>(agg, NNODES * HIDC);
        k_eagg<<<gEH, T, 0, stream>>>(ei, enorm, hlin, agg);
        k_post<<<gNH, T, 0, stream>>>(agg, hlin, snorm, b0 + (size_t)r * HIDC, feat, 0, h1bf);

        // layer 2: hlin = h1 @ W1
        k_gemm_bf16wmma<<<gGEMM, T, 0, stream>>>(h1bf, Bt1, HIDC, hlin, NNODES);
        k_zero<<<gNH, T, 0, stream>>>(agg, NNODES * HIDC);
        k_eagg<<<gEH, T, 0, stream>>>(ei, enorm, hlin, agg);
        k_post<<<gNH, T, 0, stream>>>(agg, hlin, snorm, b1 + (size_t)r * HIDC, feat, HIDC,
                                      (__bf16*)nullptr);

        // layer 3 (64 -> 1)
        k_dot3 <<<gN, T, 0, stream>>>(feat, W2 + (size_t)r * HIDC, hl3);
        k_zero <<<gN, T, 0, stream>>>(agg1, NNODES);
        k_eagg1<<<gE, T, 0, stream>>>(ei, enorm, hl3, agg1);
        k_post3<<<gN, T, 0, stream>>>(agg1, hl3, snorm, b2 + r, feat);

        // sort-pool + CNN head, accumulate into out
        k_topk<<<NGROUP, 32, 0, stream>>>(feat, glist, gstart, pooled);
        k_cnn <<<NGROUP, 128, 0, stream>>>(pooled, c1w, c1b, c2w, c2b, out);
    }
}
